// TemporalRelationGraph_5454608466452
// MI455X (gfx1250) — matrix operand order repaired and checked by reading, not compiled, not run on hardware
//
#include <hip/hip_runtime.h>
#include <math.h>

// ---------------------------------------------------------------------------
// Shapes / constants from the reference
//   B=256 SEQ=8 K=3 SD=256 NB=512 NC=174 NOBJ=10 D=(K+1)*SD=1024
//   P columns: scale4 j0..3 | scale3 j0..2 | scale2 j0..1 | scale1 j0  = 5120
//   Hcat columns: 4 scales x 512 = 2048 ; W2cat padded to 256 cols
// ---------------------------------------------------------------------------
#define BATCH_  256
#define SEQ_    8
#define SD_     256
#define D_      1024
#define NB_     512
#define NC_     174
#define NOBJ_   10
#define MROWS   2048            // B*SEQ
#define PCOLS   5120
#define HCOLS   2048
#define W2LD    256

typedef __attribute__((ext_vector_type(8)))  __bf16 bf16x8;
typedef __attribute__((ext_vector_type(16))) __bf16 bf16x16;
typedef __attribute__((ext_vector_type(8)))  float  floatx8;

union FragU { bf16x16 v; bf16x8 h[2]; };

// Plan tables (is_test is irrelevant: SUB=10 > len(SCALES) makes plans equal)
__constant__ unsigned char GCNT[4][8] = {
  {0,0,0,1,2,3,4,2},   // scale 4
  {0,0,1,2,3,2,3,4},   // scale 3
  {0,1,2,2,3,3,4,4},   // scale 2
  {1,2,3,4,5,6,7,8}    // scale 1
};
__constant__ signed char GT[4][8][8][4] = {
  { // scale 4
    {}, {}, {},
    { {0,1,2,3} },
    { {0,1,2,3},{1,2,3,4} },
    { {0,1,2,3},{1,2,3,4},{2,3,4,5} },
    { {0,1,2,3},{1,2,3,4},{2,3,4,5},{3,4,5,6} },
    { {0,2,4,6},{1,3,5,7} }
  },
  { // scale 3
    {}, {},
    { {0,1,2} },
    { {0,1,2},{1,2,3} },
    { {0,1,2},{1,2,3},{2,3,4} },
    { {0,2,4},{1,3,5} },
    { {0,2,4},{1,3,5},{2,4,6} },
    { {0,2,4},{1,3,5},{2,4,6},{3,5,7} }
  },
  { // scale 2
    {},
    { {0,1} },
    { {0,1},{1,2} },
    { {0,2},{1,3} },
    { {0,2},{1,3},{2,4} },
    { {0,3},{1,4},{2,5} },
    { {0,3},{1,4},{2,5},{3,6} },
    { {0,4},{1,5},{2,6},{3,7} }
  },
  { // scale 1
    { {0} },
    { {0},{1} },
    { {0},{1},{2} },
    { {0},{1},{2},{3} },
    { {0},{1},{2},{3},{4} },
    { {0},{1},{2},{3},{4},{5} },
    { {0},{1},{2},{3},{4},{5},{6} },
    { {0},{1},{2},{3},{4},{5},{6},{7} }
  }
};
__constant__ float RCNTF[8][4] = {   // groups per (row, scale-idx) -> bias multiplier
  {0,0,0,1},{0,0,1,2},{0,1,2,3},{1,2,2,4},{2,3,3,5},{3,2,3,6},{4,3,4,7},{2,4,4,8}
};
__constant__ int SEGBASE[4] = {0, 2048, 3584, 4608};

// ---------------------------------------------------------------------------
// K0a: build W1cat (1024 x 5120) as hi/lo bf16 split of the fp32 weights
// ---------------------------------------------------------------------------
__global__ __launch_bounds__(256) void k0_w1(
    const float* __restrict__ w14, const float* __restrict__ w13,
    const float* __restrict__ w12, const float* __restrict__ w11,
    __bf16* __restrict__ W1h, __bf16* __restrict__ W1l) {
  int e = blockIdx.x * 256 + threadIdx.x;          // e = k*5120 + col
  int k = e / PCOLS, col = e - k * PCOLS;
  int s_idx, rel;
  if      (col < 2048) { s_idx = 0; rel = col; }
  else if (col < 3584) { s_idx = 1; rel = col - 2048; }
  else if (col < 4608) { s_idx = 2; rel = col - 3584; }
  else                 { s_idx = 3; rel = col - 4608; }
  int j = rel >> 9, n = rel & 511;
  const float* w = (s_idx == 0) ? w14 : (s_idx == 1) ? w13 : (s_idx == 2) ? w12 : w11;
  float v = w[(size_t)(j * D_ + k) * NB_ + n];
  __bf16 hi = (__bf16)v;
  W1h[e] = hi;
  W1l[e] = (__bf16)(v - (float)hi);
}

// ---------------------------------------------------------------------------
// K0b: build W2cat (2048 x 256, cols >= 174 zero) hi/lo bf16
// ---------------------------------------------------------------------------
__global__ __launch_bounds__(256) void k0_w2(
    const float* __restrict__ w24, const float* __restrict__ w23,
    const float* __restrict__ w22, const float* __restrict__ w21,
    __bf16* __restrict__ W2h, __bf16* __restrict__ W2l) {
  int e = blockIdx.x * 256 + threadIdx.x;          // e = k*256 + n
  int k = e >> 8, n = e & 255;
  int s_idx = k >> 9, nb = k & 511;
  const float* w = (s_idx == 0) ? w24 : (s_idx == 1) ? w23 : (s_idx == 2) ? w22 : w21;
  float v = (n < NC_) ? w[(size_t)nb * NC_ + n] : 0.f;
  __bf16 hi = (__bf16)v;
  W2h[e] = hi;
  W2l[e] = (__bf16)(v - (float)hi);
}

// ---------------------------------------------------------------------------
// K1: sig2[b*3+k] = lambda_max( diff diff^T ), diff = flat[b] - flat[b][k]
//     flat[b][i][d] == init[(b*8)*2560 + i*256 + d]   (i in 0..79)
//     LDS: 80x64 f32 diff chunk (+1 pad -> conflict free) + 80x80 Gram,
//     5x5 register blocking, then 48 power iterations.
// ---------------------------------------------------------------------------
__global__ __launch_bounds__(256) void k1_sigma(
    const float* __restrict__ init, float* __restrict__ sig2) {
  __shared__ float sd[80 * 65];
  __shared__ float G[80 * 80];
  __shared__ float vv[80], uu[80];
  __shared__ float red;
  int bx = blockIdx.x;
  int b = bx / 3, kk = bx - 3 * b;
  int tid = threadIdx.x;
  const float* base = init + (size_t)b * SEQ_ * (NOBJ_ * SD_);
  const float* cen  = base + kk * SD_;

  float acc[25];
#pragma unroll
  for (int q = 0; q < 25; ++q) acc[q] = 0.f;
  int i0 = (tid >> 4) * 5, j0 = (tid & 15) * 5;

  for (int c = 0; c < 4; ++c) {
    for (int e = tid; e < 80 * 64; e += 256) {
      int i = e >> 6, d = e & 63;
      int gd = c * 64 + d;
      sd[i * 65 + d] = base[i * SD_ + gd] - cen[gd];
    }
    __syncthreads();
    for (int d = 0; d < 64; ++d) {
      float ai[5], aj[5];
#pragma unroll
      for (int r = 0; r < 5; ++r) { ai[r] = sd[(i0 + r) * 65 + d]; aj[r] = sd[(j0 + r) * 65 + d]; }
#pragma unroll
      for (int r = 0; r < 5; ++r)
#pragma unroll
        for (int cj = 0; cj < 5; ++cj) acc[r * 5 + cj] += ai[r] * aj[cj];
    }
    __syncthreads();
  }
#pragma unroll
  for (int r = 0; r < 5; ++r)
#pragma unroll
    for (int cj = 0; cj < 5; ++cj) G[(i0 + r) * 80 + (j0 + cj)] = acc[r * 5 + cj];
  if (tid < 80) vv[tid] = 1.0f;
  __syncthreads();

  float lam = 0.f;
  for (int it = 0; it < 48; ++it) {
    if (tid < 80) {
      float s = 0.f;
      for (int j = 0; j < 80; ++j) s += G[tid * 80 + j] * vv[j];
      uu[tid] = s;
    }
    __syncthreads();
    if (tid == 0) {
      float s = 0.f;
      for (int j = 0; j < 80; ++j) s += uu[j] * uu[j];
      red = sqrtf(s);
    }
    __syncthreads();
    float nrm = red;
    if (tid < 80) vv[tid] = uu[tid] / fmaxf(nrm, 1e-30f);
    lam = nrm;                      // ||G v_unit|| -> lambda_max = sigma_max^2
    __syncthreads();
  }
  if (tid == 0) sig2[bx] = lam;
}

// ---------------------------------------------------------------------------
// K2: build X = relu(new_input) (2048 x 1024) as hi/lo bf16.
//     cols 0..767 = var (k-major), cols 768..1023 = graph_out
// ---------------------------------------------------------------------------
static __device__ inline void st_split(__bf16* H, __bf16* L, size_t i, float v) {
  __bf16 hi = (__bf16)v;
  H[i] = hi;
  L[i] = (__bf16)(v - (float)hi);
}

__global__ __launch_bounds__(256) void k2_build(
    const float* __restrict__ go, const float* __restrict__ init,
    const float* __restrict__ sig2, __bf16* __restrict__ Xh, __bf16* __restrict__ Xl) {
  int bx = blockIdx.x;               // row = b*8 + t
  int b = bx >> 3;
  int d = threadIdx.x;

  float a0 = -0.1f * sig2[b * 3 + 0];
  float a1 = -0.1f * sig2[b * 3 + 1];
  float a2 = -0.1f * sig2[b * 3 + 2];
  float mx = fmaxf(a0, fmaxf(a1, a2));
  float e0 = __expf(a0 - mx), e1 = __expf(a1 - mx), e2 = __expf(a2 - mx);
  float inv = 1.f / (e0 + e1 + e2);
  float w0 = e0 * inv, w1 = e1 * inv, w2 = e2 * inv;

  const float* rowp = init + (size_t)bx * (NOBJ_ * SD_);
  float xs = 0.f;
#pragma unroll
  for (int o = 0; o < NOBJ_; ++o) xs += rowp[o * SD_ + d];

  const float* cb = init + (size_t)(bx & ~7) * (NOBJ_ * SD_);   // t=0 of this b
  float c0 = cb[0 * SD_ + d], c1 = cb[1 * SD_ + d], c2 = cb[2 * SD_ + d];

  size_t xb = (size_t)bx * D_;
  st_split(Xh, Xl, xb + 0 * SD_ + d, fmaxf(w0 * (xs - 10.f * c0), 0.f));
  st_split(Xh, Xl, xb + 1 * SD_ + d, fmaxf(w1 * (xs - 10.f * c1), 0.f));
  st_split(Xh, Xl, xb + 2 * SD_ + d, fmaxf(w2 * (xs - 10.f * c2), 0.f));
  st_split(Xh, Xl, xb + 3 * SD_ + d, fmaxf(go[(size_t)bx * SD_ + d], 0.f));
}

// ---------------------------------------------------------------------------
// WMMA GEMM core (wave32). Per-wave 32x32 tile = 2x2 v_wmma_f32_16x16x32_bf16
// tiles; split-bf16 product (Ah*Bh + Ah*Bl + Al*Bh) for ~fp32 accuracy.
// Fragment layouts per ISA 7.12.2:
//   A 16x32: lane = M row (lane&15); K halves {0..7,16..23} lanes<16,
//            {8..15,24..31} lanes>=16 -> two 16B loads at k+kh and k+kh+16.
//   B 32x16: lane = N col (lane&15); K halves same as A -> strided scalar loads.
//   C/D:     8 VGPRs; M = g (+8 for lanes>=16), N = lane&15.
// ---------------------------------------------------------------------------
static __device__ inline bf16x16 load_a(const __bf16* A, int lda, int row, int kb) {
  FragU f;
  const __bf16* p = A + (size_t)row * lda + kb;
  f.h[0] = *reinterpret_cast<const bf16x8*>(p);
  f.h[1] = *reinterpret_cast<const bf16x8*>(p + 16);
  return f.v;
}
static __device__ inline bf16x16 load_b(const __bf16* B, int ldb, int krow, int col) {
  FragU f;
  const __bf16* p = B + (size_t)krow * ldb + col;
#pragma unroll
  for (int s = 0; s < 16; ++s) f.v[s] = p[(size_t)(s < 8 ? s : s + 8) * ldb];
  return f.v;
}

static __device__ inline void gemm_core(
    const __bf16* __restrict__ Ah, const __bf16* __restrict__ Al,
    const __bf16* __restrict__ Bh, const __bf16* __restrict__ Bl,
    int lda, int ldb, int K, int m0, int n0, int lane, floatx8 (&acc)[2][2]) {
  int kh = (lane & 16) ? 8 : 0;
  int am = lane & 15;
  for (int k = 0; k < K; k += 32) {
    bf16x16 a_h[2], a_l[2], b_h[2], b_l[2];
#pragma unroll
    for (int mt = 0; mt < 2; ++mt) {
      int row = m0 + mt * 16 + am;
      a_h[mt] = load_a(Ah, lda, row, k + kh);
      a_l[mt] = load_a(Al, lda, row, k + kh);
    }
    int brow = k + kh;
#pragma unroll
    for (int nt = 0; nt < 2; ++nt) {
      int col = n0 + nt * 16 + am;
      b_h[nt] = load_b(Bh, ldb, brow, col);
      b_l[nt] = load_b(Bl, ldb, brow, col);
    }
#pragma unroll
    for (int mt = 0; mt < 2; ++mt)
#pragma unroll
      for (int nt = 0; nt < 2; ++nt) {
        acc[mt][nt] = __builtin_amdgcn_wmma_f32_16x16x32_bf16(
            false, a_h[mt], false, b_h[nt], (short)0, acc[mt][nt], false, false);
        acc[mt][nt] = __builtin_amdgcn_wmma_f32_16x16x32_bf16(
            false, a_h[mt], false, b_l[nt], (short)0, acc[mt][nt], false, false);
        acc[mt][nt] = __builtin_amdgcn_wmma_f32_16x16x32_bf16(
            false, a_l[mt], false, b_h[nt], (short)0, acc[mt][nt], false, false);
      }
  }
}

// GEMM1: P(2048 x 5120) = X(2048x1024) @ W1cat(1024x5120)
__global__ __launch_bounds__(256) void gemm1(
    const __bf16* __restrict__ Xh, const __bf16* __restrict__ Xl,
    const __bf16* __restrict__ W1h, const __bf16* __restrict__ W1l,
    float* __restrict__ P) {
  int lane = threadIdx.x & 31, w = threadIdx.x >> 5;
  int m0 = blockIdx.y * 64 + (w >> 2) * 32;
  int n0 = blockIdx.x * 128 + (w & 3) * 32;
  floatx8 z = {0.f, 0.f, 0.f, 0.f, 0.f, 0.f, 0.f, 0.f};
  floatx8 acc[2][2] = {{z, z}, {z, z}};
  gemm_core(Xh, Xl, W1h, W1l, D_, PCOLS, D_, m0, n0, lane, acc);
  int rb = (lane & 16) ? 8 : 0, cn = lane & 15;
#pragma unroll
  for (int mt = 0; mt < 2; ++mt)
#pragma unroll
    for (int nt = 0; nt < 2; ++nt) {
      int col = n0 + nt * 16 + cn;
      int row = m0 + mt * 16 + rb;
#pragma unroll
      for (int g = 0; g < 8; ++g)
        P[(size_t)(row + g) * PCOLS + col] = acc[mt][nt][g];
    }
}

// ---------------------------------------------------------------------------
// K3: Hcat[m, s*512+n] = sum_groups relu( b1_s[n] + sum_j P[b8+t_j, seg+j*512+n] )
// ---------------------------------------------------------------------------
__global__ __launch_bounds__(256) void k3_hcat(
    const float* __restrict__ P,
    const float* __restrict__ b14, const float* __restrict__ b13,
    const float* __restrict__ b12, const float* __restrict__ b11,
    __bf16* __restrict__ Hh, __bf16* __restrict__ Hl) {
  int m = blockIdx.x, s_idx = blockIdx.y, tid = threadIdx.x;
  int row = m & 7, base = m - row;
  const float* b1 = (s_idx == 0) ? b14 : (s_idx == 1) ? b13 : (s_idx == 2) ? b12 : b11;
  int s = 4 - s_idx;
  int seg = SEGBASE[s_idx];
  int gc = GCNT[s_idx][row];
#pragma unroll
  for (int nn = 0; nn < 2; ++nn) {
    int n = tid + nn * 256;
    float bias = b1[n];
    float acc = 0.f;
    for (int g = 0; g < gc; ++g) {
      float v = bias;
      for (int j = 0; j < s; ++j) {
        int t = GT[s_idx][row][g][j];
        v += P[(size_t)(base + t) * PCOLS + seg + j * NB_ + n];
      }
      acc += fmaxf(v, 0.f);
    }
    st_split(Hh, Hl, (size_t)m * HCOLS + s_idx * NB_ + n, acc);
  }
}

// GEMM2: out(2048 x 174) = Hcat(2048x2048) @ W2cat(2048x256) + row-count * b2
__global__ __launch_bounds__(256) void gemm2(
    const __bf16* __restrict__ Hh, const __bf16* __restrict__ Hl,
    const __bf16* __restrict__ W2h, const __bf16* __restrict__ W2l,
    const float* __restrict__ b24, const float* __restrict__ b23,
    const float* __restrict__ b22, const float* __restrict__ b21,
    float* __restrict__ out) {
  int lane = threadIdx.x & 31, w = threadIdx.x >> 5;
  int m0 = blockIdx.y * 64 + (w >> 2) * 32;
  int n0 = blockIdx.x * 128 + (w & 3) * 32;
  floatx8 z = {0.f, 0.f, 0.f, 0.f, 0.f, 0.f, 0.f, 0.f};
  floatx8 acc[2][2] = {{z, z}, {z, z}};
  gemm_core(Hh, Hl, W2h, W2l, HCOLS, W2LD, HCOLS, m0, n0, lane, acc);
  int rb = (lane & 16) ? 8 : 0, cn = lane & 15;
#pragma unroll
  for (int mt = 0; mt < 2; ++mt)
#pragma unroll
    for (int nt = 0; nt < 2; ++nt) {
      int col = n0 + nt * 16 + cn;
      if (col >= NC_) continue;
      float bv4 = b24[col], bv3 = b23[col], bv2 = b22[col], bv1 = b21[col];
      int row = m0 + mt * 16 + rb;
#pragma unroll
      for (int g = 0; g < 8; ++g) {
        int r = row + g, sr = r & 7;
        float bias = RCNTF[sr][0] * bv4 + RCNTF[sr][1] * bv3 +
                     RCNTF[sr][2] * bv2 + RCNTF[sr][3] * bv1;
        out[(size_t)r * NC_ + col] = acc[mt][nt][g] + bias;
      }
    }
}

// ---------------------------------------------------------------------------
extern "C" void kernel_launch(void* const* d_in, const int* in_sizes, int n_in,
                              void* d_out, int out_size, void* d_ws, size_t ws_size,
                              hipStream_t stream) {
  const float* graph_out = (const float*)d_in[0];
  const float* init      = (const float*)d_in[1];
  // d_in[2] = is_test : plan is identical for train/test (SUB=10 >= len(SCALES))
  const float* w1_4 = (const float*)d_in[3];
  const float* b1_4 = (const float*)d_in[4];
  const float* w2_4 = (const float*)d_in[5];
  const float* b2_4 = (const float*)d_in[6];
  const float* w1_3 = (const float*)d_in[7];
  const float* b1_3 = (const float*)d_in[8];
  const float* w2_3 = (const float*)d_in[9];
  const float* b2_3 = (const float*)d_in[10];
  const float* w1_2 = (const float*)d_in[11];
  const float* b1_2 = (const float*)d_in[12];
  const float* w2_2 = (const float*)d_in[13];
  const float* b2_2 = (const float*)d_in[14];
  const float* w1_1 = (const float*)d_in[15];
  const float* b1_1 = (const float*)d_in[16];
  const float* w2_1 = (const float*)d_in[17];
  const float* b2_1 = (const float*)d_in[18];
  float* out = (float*)d_out;

  // Workspace carve-up (~90 MB total, all 256B aligned)
  char* ws = (char*)d_ws;
  size_t off = 0;
  auto carve = [&](size_t bytes) {
    size_t o = off;
    off = (off + bytes + 255) & ~(size_t)255;
    return o;
  };
  const size_t X_EL  = (size_t)MROWS * D_;
  const size_t W1_EL = (size_t)D_ * PCOLS;
  const size_t W2_EL = (size_t)HCOLS * W2LD;
  const size_t P_EL  = (size_t)MROWS * PCOLS;
  const size_t H_EL  = (size_t)MROWS * HCOLS;
  __bf16* Xh  = (__bf16*)(ws + carve(X_EL * 2));
  __bf16* Xl  = (__bf16*)(ws + carve(X_EL * 2));
  __bf16* W1h = (__bf16*)(ws + carve(W1_EL * 2));
  __bf16* W1l = (__bf16*)(ws + carve(W1_EL * 2));
  __bf16* W2h = (__bf16*)(ws + carve(W2_EL * 2));
  __bf16* W2l = (__bf16*)(ws + carve(W2_EL * 2));
  float*  P   = (float*)(ws + carve(P_EL * 4));
  __bf16* Hh  = (__bf16*)(ws + carve(H_EL * 2));
  __bf16* Hl  = (__bf16*)(ws + carve(H_EL * 2));
  float*  sg2 = (float*)(ws + carve(768 * 4));
  (void)ws_size; (void)in_sizes; (void)n_in; (void)out_size;

  // Weight conversion (must rerun every call: no cross-call state allowed)
  k0_w1<<<(int)(W1_EL / 256), 256, 0, stream>>>(w1_4, w1_3, w1_2, w1_1, W1h, W1l);
  k0_w2<<<(int)(W2_EL / 256), 256, 0, stream>>>(w2_4, w2_3, w2_2, w2_1, W2h, W2l);

  // sigma^2 via Gram + power iteration
  k1_sigma<<<BATCH_ * 3, 256, 0, stream>>>(init, sg2);

  // Build relu(new_input) as split-bf16
  k2_build<<<MROWS, 256, 0, stream>>>(graph_out, init, sg2, Xh, Xl);

  // GEMM1: all 10 (scale, j) weight blocks at once -> partials P
  gemm1<<<dim3(PCOLS / 128, MROWS / 64), 256, 0, stream>>>(Xh, Xl, W1h, W1l, P);

  // Combine partials per plan group -> Hcat
  k3_hcat<<<dim3(MROWS, 4), 256, 0, stream>>>(P, b1_4, b1_3, b1_2, b1_1, Hh, Hl);

  // GEMM2: Hcat @ W2cat + per-row group-count * b2
  gemm2<<<dim3(2, MROWS / 64), 256, 0, stream>>>(Hh, Hl, W2h, W2l,
                                                 b2_4, b2_3, b2_2, b2_1, out);
}
